// EdgeGatedGraphConv_25082609008766
// MI455X (gfx1250) — matrix operand order repaired and checked by reading, not compiled, not run on hardware
//
#include <hip/hip_runtime.h>
#include <hip/hip_bf16.h>

// EdgeGatedGraphConv for MI455X (gfx1250), wave32 + WMMA bf16 + TDM weight staging.
// ws layout (floats): Q[N*128] | K[N*128] | V[N*128] | AGG[N*128]  then
//                     bf16 weights: 6 x 128x128 __bf16 (= 102.4 MB + 0.2 MB)

#define NNODES 50000
#define NEDGES 400000
#define FDIM   128
#define NHEAD  4
#define CDIM   32

#define INV_SCALE 0.17677669529663687f   // 1/sqrt(32)
#define LN_EPS    1e-5f

// LDS weight row stride (bf16 elems): 128 + 8 pad -> 272B rows, bank-conflict-free
#define WSTRIDE 136
#define WBYTES  (FDIM * WSTRIDE * 2)     // bytes per staged weight matrix

typedef __attribute__((ext_vector_type(16))) __bf16 v16bf;
typedef __attribute__((ext_vector_type(8)))  __bf16 v8bf;
typedef __attribute__((ext_vector_type(8)))  float  v8f;

typedef __attribute__((ext_vector_type(4))) unsigned int u32x4;
typedef __attribute__((ext_vector_type(4))) int          i32x4;
typedef __attribute__((ext_vector_type(8))) int          i32x8;

// ---- Tensor Data Mover: stage one 128x128 bf16 weight (row-major) into LDS
// with +16B padding per 256B row (descriptor per CDNA5 ISA 08_async_tensor).
// Toolchain uses the 6-arg builtin form: (g0, g1, g2, g3, g4, cpol).
__device__ __forceinline__ void tdm_stage_weight(const __bf16* gsrc,
                                                 __bf16* lds_dst) {
  const unsigned long long ga = (unsigned long long)(uintptr_t)gsrc;
  const unsigned lds_off = (unsigned)(uintptr_t)lds_dst;  // low 32 = LDS byte addr
  u32x4 g0;
  g0[0] = 1u;                                   // count=1, no gather/iterate
  g0[1] = lds_off;                              // lds_addr [63:32]
  g0[2] = (unsigned)ga;                         // global_addr [95:64]
  g0[3] = (unsigned)((ga >> 32) & 0x1FFFFFFull) // global_addr [120:96]
          | (2u << 30);                         // type=2 (image) [127:126]
  i32x8 g1;
  g1[0] = (1 << 16)                             // data_size = 2 bytes
        | (1 << 20)                             // pad_enable
        | (5 << 22)                             // pad_interval: 64 DW = 256B
        | (3 << 25);                            // pad_amount:   4 DW = 16B
  g1[1] = (FDIM << 16);                         // tensor_dim0 = 128
  g1[2] = (FDIM << 16);                         // tensor_dim1 = 128
  g1[3] = (FDIM << 16);                         // tile_dim0   = 128
  g1[4] = FDIM;                                 // tile_dim1   = 128 (tile_dim2=0)
  g1[5] = FDIM;                                 // tensor_dim0_stride = 128
  g1[6] = 0;
  g1[7] = 0;
  i32x4 gz;
  gz[0] = 0; gz[1] = 0; gz[2] = 0; gz[3] = 0;   // 2-D tile: groups 2/3 unused
  i32x8 gz8;
  gz8[0] = 0; gz8[1] = 0; gz8[2] = 0; gz8[3] = 0;
  gz8[4] = 0; gz8[5] = 0; gz8[6] = 0; gz8[7] = 0;
  __builtin_amdgcn_tensor_load_to_lds(g0, g1, gz, gz, gz8, 0);
}

// ---- WMMA fragment loaders (layouts per CDNA5 ISA 7.12.2, wave32) ----------
// A (16x32 bf16, M x K) from f32 global: lane L in [0,16): row=L,
// K = {kb..kb+7, kb+16..kb+23}; lanes 16..31: kb offset +8 / +24.
__device__ __forceinline__ v16bf load_a_frag(const float* __restrict__ base,
                                             int kstep, int lane) {
  const int row = lane & 15;
  const int kb  = kstep * 32 + ((lane >> 4) << 3);
  const float* p = base + (size_t)row * FDIM + kb;
  v16bf a;
#pragma unroll
  for (int j = 0; j < 8; ++j) a[j]     = (__bf16)p[j];
#pragma unroll
  for (int j = 0; j < 8; ++j) a[8 + j] = (__bf16)p[16 + j];
  return a;
}

// B (32x16 bf16, K x N), B[k][n] = W[n][k], W staged in LDS with WSTRIDE rows:
// lanes 0..15: col n0+lane, K={kb..kb+7, kb+16..kb+23}; lanes 16..31: K += 8.
__device__ __forceinline__ v16bf load_b_frag_lds(const __bf16* Wl, int n0,
                                                 int kstep, int lane) {
  const int col = lane & 15;
  const int kb  = kstep * 32 + ((lane >> 4) << 3);
  const __bf16* p = Wl + (n0 + col) * WSTRIDE + kb;   // 16B-aligned
  const v8bf lo = *(const v8bf*)p;
  const v8bf hi = *(const v8bf*)(p + 16);
  v16bf b;
#pragma unroll
  for (int j = 0; j < 8; ++j) { b[j] = lo[j]; b[8 + j] = hi[j]; }
  return b;
}

__device__ __forceinline__ v8f wmma_bf16(v16bf a, v16bf b, v8f c) {
  return __builtin_amdgcn_wmma_f32_16x16x32_bf16(false, a, false, b,
                                                 (short)0, c, false, false);
}

// ---------------------------------------------------------------------------
__global__ void __launch_bounds__(256)
zero_kernel(float4* __restrict__ p, int n4) {
  for (int i = blockIdx.x * blockDim.x + threadIdx.x; i < n4;
       i += gridDim.x * blockDim.x)
    p[i] = make_float4(0.f, 0.f, 0.f, 0.f);
}

// Convert the 6 weight matrices to bf16 once (row-major, unpadded, in ws).
__global__ void __launch_bounds__(256)
prep_weights_kernel(const float* __restrict__ Wq, const float* __restrict__ Wk,
                    const float* __restrict__ Wv, const float* __restrict__ We,
                    const float* __restrict__ Wev, const float* __restrict__ Wout,
                    __bf16* __restrict__ out) {
  const int m = blockIdx.y;
  const float* srcs[6] = {Wq, Wk, Wv, We, Wev, Wout};
  const float* s = srcs[m];
  const int i = blockIdx.x * blockDim.x + threadIdx.x;   // 0..16383
  out[m * (FDIM * FDIM) + i] = (__bf16)s[i];
}

// Q/K/V projections: one wave per 16-node tile, A loaded once, 3 GEMMs.
__global__ void __launch_bounds__(256)
node_proj_kernel(const float* __restrict__ node_feats,
                 const __bf16* __restrict__ WB,       // bf16 weights in ws
                 const float* __restrict__ bq, const float* __restrict__ bk,
                 const float* __restrict__ bv,
                 float* __restrict__ Q, float* __restrict__ K,
                 float* __restrict__ V) {
  extern __shared__ __bf16 smem[];                     // 3 * 128*136 bf16
  if (threadIdx.x == 0) {
    tdm_stage_weight(WB + 0 * FDIM * FDIM, smem + 0 * FDIM * WSTRIDE);
    tdm_stage_weight(WB + 1 * FDIM * FDIM, smem + 1 * FDIM * WSTRIDE);
    tdm_stage_weight(WB + 2 * FDIM * FDIM, smem + 2 * FDIM * WSTRIDE);
    __builtin_amdgcn_s_wait_tensorcnt(0);
  }
  __syncthreads();

  const int lane = threadIdx.x & 31;
  const int wave = threadIdx.x >> 5;
  const int tile = blockIdx.x * 8 + wave;
  if (tile >= NNODES / 16) return;
  const int base = tile * 16;

  v16bf a[4];
#pragma unroll
  for (int ks = 0; ks < 4; ++ks)
    a[ks] = load_a_frag(node_feats + (size_t)base * FDIM, ks, lane);

  const int lcol  = lane & 15;
  const int rbase = (lane >> 4) << 3;

  const float* bs[3] = {bq, bk, bv};
  float*       Os[3] = {Q, K, V};
#pragma unroll
  for (int w = 0; w < 3; ++w) {
    const __bf16* Wl = smem + w * FDIM * WSTRIDE;
#pragma unroll
    for (int nt = 0; nt < 8; ++nt) {
      const int col = nt * 16 + lcol;
      const float bb = bs[w][col];
      v8f c;
#pragma unroll
      for (int r = 0; r < 8; ++r) c[r] = bb;   // bias folded into accumulator
#pragma unroll
      for (int ks = 0; ks < 4; ++ks)
        c = wmma_bf16(a[ks], load_b_frag_lds(Wl, nt * 16, ks, lane), c);
#pragma unroll
      for (int r = 0; r < 8; ++r)
        Os[w][(size_t)(base + rbase + r) * FDIM + col] = c[r];
    }
  }
}

// Fused edge pass: e-projection GEMM, q·k scores, head-softmax, V gather +
// atomic scatter into AGG, LayerNorm+SiLU, Wev residual GEMM -> y.
__global__ void __launch_bounds__(256)
edge_kernel(const float* __restrict__ edge_feats,
            const int* __restrict__ src, const int* __restrict__ dst,
            const float* __restrict__ Q, const float* __restrict__ K,
            const float* __restrict__ V,
            const __bf16* __restrict__ WB,            // bf16 weights in ws
            const float* __restrict__ be, const float* __restrict__ bev,
            const float* __restrict__ ln_eg, const float* __restrict__ ln_eb,
            float* __restrict__ AGG, float* __restrict__ Y) {
  extern __shared__ __bf16 smem[];                    // 2 * 128*136 bf16
  if (threadIdx.x == 0) {
    tdm_stage_weight(WB + 3 * FDIM * FDIM, smem + 0 * FDIM * WSTRIDE);  // We
    tdm_stage_weight(WB + 4 * FDIM * FDIM, smem + 1 * FDIM * WSTRIDE);  // Wev
    __builtin_amdgcn_s_wait_tensorcnt(0);
  }
  __syncthreads();
  const __bf16* WeL  = smem;
  const __bf16* WevL = smem + FDIM * WSTRIDE;

  const int lane = threadIdx.x & 31;
  const int wave = threadIdx.x >> 5;
  const int tile = blockIdx.x * 8 + wave;
  if (tile >= NEDGES / 16) return;
  const int ebase = tile * 16;

  v16bf a[4];
#pragma unroll
  for (int ks = 0; ks < 4; ++ks)
    a[ks] = load_a_frag(edge_feats + (size_t)ebase * FDIM, ks, lane);

  // --- per-lane q·k scores: lane handles (edge = lane&15, heads h0 and h0+2)
  const int eL = ebase + (lane & 15);
  const int h0 = lane >> 4;                        // 0 or 1
  const int sL = src[eL], dL = dst[eL];
  const float* qp = Q + (size_t)sL * FDIM;
  const float* kp = K + (size_t)dL * FDIM;
  float s0 = 0.f, s1 = 0.f;
#pragma unroll
  for (int j = 0; j < CDIM; ++j) {
    s0 += qp[h0 * CDIM + j]       * kp[h0 * CDIM + j];
    s1 += qp[(h0 + 2) * CDIM + j] * kp[(h0 + 2) * CDIM + j];
  }

  const int lcol  = lane & 15;
  const int rbase = (lane >> 4) << 3;

  // --- m = edge_feats @ We.T + be + score/SCALE (broadcast over C)
  v8f m[8];
#pragma unroll
  for (int nt = 0; nt < 8; ++nt) {
    const int col = nt * 16 + lcol;
    const float bb = be[col];
    v8f c;
#pragma unroll
    for (int r = 0; r < 8; ++r) c[r] = bb;
#pragma unroll
    for (int ks = 0; ks < 4; ++ks)
      c = wmma_bf16(a[ks], load_b_frag_lds(WeL, nt * 16, ks, lane), c);

    const int hh     = nt >> 1;                    // head of this column tile
    const float sval = (hh < 2) ? s0 : s1;         // uniform select
    const int slbase = (hh & 1) << 4;
#pragma unroll
    for (int r = 0; r < 8; ++r) {
      const int row  = rbase + r;                  // edge row within tile
      const float sc = __shfl(sval, slbase + row, 32);
      c[r] += sc * INV_SCALE;
    }
    m[nt] = c;
  }

  // --- softmax over heads (tiles {p, p+2, p+4, p+6}), V gather, AGG scatter
#pragma unroll
  for (int r = 0; r < 8; ++r) {
    const int row = rbase + r;
    const int e2  = ebase + row;
    const int sv  = src[e2], dv = dst[e2];
#pragma unroll
    for (int p = 0; p < 2; ++p) {
      const float x0 = m[p][r],     x1 = m[p + 2][r];
      const float x2 = m[p + 4][r], x3 = m[p + 6][r];
      const float mx = fmaxf(fmaxf(x0, x1), fmaxf(x2, x3));
      const float e0 = __expf(x0 - mx), e1 = __expf(x1 - mx);
      const float e2f = __expf(x2 - mx), e3 = __expf(x3 - mx);
      const float inv = 1.f / (e0 + e1 + e2f + e3);
      const float at[4] = {e0 * inv, e1 * inv, e2f * inv, e3 * inv};
#pragma unroll
      for (int h = 0; h < 4; ++h) {
        const int colc = h * CDIM + p * 16 + lcol;
        const float vv = V[(size_t)sv * FDIM + colc];
        atomicAdd(&AGG[(size_t)dv * FDIM + colc], at[h] * vv);
      }
    }
  }

  // --- LayerNorm stats per edge row (128 cols = 8 tiles x 16-lane half-wave)
  float mu[8], rstd[8];
#pragma unroll
  for (int r = 0; r < 8; ++r) {
    float s = 0.f, s2 = 0.f;
#pragma unroll
    for (int nt = 0; nt < 8; ++nt) { const float x = m[nt][r]; s += x; s2 += x * x; }
#pragma unroll
    for (int msk = 1; msk <= 8; msk <<= 1) {
      s  += __shfl_xor(s,  msk, 32);
      s2 += __shfl_xor(s2, msk, 32);
    }
    const float mean = s * (1.f / FDIM);
    const float var  = s2 * (1.f / FDIM) - mean * mean;
    mu[r]   = mean;
    rstd[r] = rsqrtf(var + LN_EPS);
  }

  // --- y = (edge_feats @ Wev.T + bev) + silu(LN(m))
#pragma unroll
  for (int nt = 0; nt < 8; ++nt) {
    const int col = nt * 16 + lcol;
    const float bb = bev[col];
    v8f c;
#pragma unroll
    for (int r = 0; r < 8; ++r) c[r] = bb;
#pragma unroll
    for (int ks = 0; ks < 4; ++ks)
      c = wmma_bf16(a[ks], load_b_frag_lds(WevL, nt * 16, ks, lane), c);

    const float g = ln_eg[col], bt = ln_eb[col];
#pragma unroll
    for (int r = 0; r < 8; ++r) {
      const float ln = (m[nt][r] - mu[r]) * rstd[r] * g + bt;
      const float sl = ln / (1.f + __expf(-ln));   // silu
      Y[(size_t)(ebase + rbase + r) * FDIM + col] = c[r] + sl;
    }
  }
}

// Final node pass: x = (node_feats @ Wout.T + bout) + silu(LN(AGG))
__global__ void __launch_bounds__(256)
node_final_kernel(const float* __restrict__ node_feats,
                  const float* __restrict__ AGG,
                  const __bf16* __restrict__ WB,      // bf16 weights in ws
                  const float* __restrict__ bout,
                  const float* __restrict__ ln_ng, const float* __restrict__ ln_nb,
                  float* __restrict__ X) {
  extern __shared__ __bf16 smem[];                    // 128*136 bf16
  if (threadIdx.x == 0) {
    tdm_stage_weight(WB + 5 * FDIM * FDIM, smem);     // Wout
    __builtin_amdgcn_s_wait_tensorcnt(0);
  }
  __syncthreads();

  const int lane = threadIdx.x & 31;
  const int wave = threadIdx.x >> 5;
  const int tile = blockIdx.x * 8 + wave;
  if (tile >= NNODES / 16) return;
  const int base = tile * 16;

  v16bf a[4];
#pragma unroll
  for (int ks = 0; ks < 4; ++ks)
    a[ks] = load_a_frag(node_feats + (size_t)base * FDIM, ks, lane);

  const int lcol  = lane & 15;
  const int rbase = (lane >> 4) << 3;

  v8f m[8];
#pragma unroll
  for (int nt = 0; nt < 8; ++nt)
#pragma unroll
    for (int r = 0; r < 8; ++r)
      m[nt][r] = AGG[(size_t)(base + rbase + r) * FDIM + nt * 16 + lcol];

  float mu[8], rstd[8];
#pragma unroll
  for (int r = 0; r < 8; ++r) {
    float s = 0.f, s2 = 0.f;
#pragma unroll
    for (int nt = 0; nt < 8; ++nt) { const float x = m[nt][r]; s += x; s2 += x * x; }
#pragma unroll
    for (int msk = 1; msk <= 8; msk <<= 1) {
      s  += __shfl_xor(s,  msk, 32);
      s2 += __shfl_xor(s2, msk, 32);
    }
    const float mean = s * (1.f / FDIM);
    const float var  = s2 * (1.f / FDIM) - mean * mean;
    mu[r]   = mean;
    rstd[r] = rsqrtf(var + LN_EPS);
  }

#pragma unroll
  for (int nt = 0; nt < 8; ++nt) {
    const int col = nt * 16 + lcol;
    const float bb = bout[col];
    v8f c;
#pragma unroll
    for (int r = 0; r < 8; ++r) c[r] = bb;
#pragma unroll
    for (int ks = 0; ks < 4; ++ks)
      c = wmma_bf16(a[ks], load_b_frag_lds(smem, nt * 16, ks, lane), c);

    const float g = ln_ng[col], bt = ln_nb[col];
#pragma unroll
    for (int r = 0; r < 8; ++r) {
      const float ln = (m[nt][r] - mu[r]) * rstd[r] * g + bt;
      const float sl = ln / (1.f + __expf(-ln));
      X[(size_t)(base + rbase + r) * FDIM + col] = c[r] + sl;
    }
  }
}

// ---------------------------------------------------------------------------
extern "C" void kernel_launch(void* const* d_in, const int* in_sizes, int n_in,
                              void* d_out, int out_size, void* d_ws,
                              size_t ws_size, hipStream_t stream) {
  const float* node_feats = (const float*)d_in[0];
  const float* edge_feats = (const float*)d_in[1];
  const int*   src        = (const int*)d_in[2];
  const int*   dst        = (const int*)d_in[3];
  const float* Wq   = (const float*)d_in[4];
  const float* bq   = (const float*)d_in[5];
  const float* Wk   = (const float*)d_in[6];
  const float* bk   = (const float*)d_in[7];
  const float* Wv   = (const float*)d_in[8];
  const float* bv   = (const float*)d_in[9];
  const float* We   = (const float*)d_in[10];
  const float* be   = (const float*)d_in[11];
  const float* Wev  = (const float*)d_in[12];
  const float* bev  = (const float*)d_in[13];
  const float* Wout = (const float*)d_in[14];
  const float* bout = (const float*)d_in[15];
  const float* ln_ng = (const float*)d_in[16];
  const float* ln_nb = (const float*)d_in[17];
  const float* ln_eg = (const float*)d_in[18];
  const float* ln_eb = (const float*)d_in[19];

  float* ws  = (float*)d_ws;
  float* Q   = ws;
  float* K   = ws + (size_t)NNODES * FDIM;
  float* V   = ws + (size_t)2 * NNODES * FDIM;
  float* AGG = ws + (size_t)3 * NNODES * FDIM;
  __bf16* WB = (__bf16*)(ws + (size_t)4 * NNODES * FDIM);  // 6 x 128x128 bf16

  float* X = (float*)d_out;                       // [N,128]
  float* Y = X + (size_t)NNODES * FDIM;           // [E,128]

  zero_kernel<<<1024, 256, 0, stream>>>((float4*)AGG, NNODES * FDIM / 4);
  prep_weights_kernel<<<dim3(FDIM * FDIM / 256, 6), 256, 0, stream>>>(
      Wq, Wk, Wv, We, Wev, Wout, WB);

  const int node_tiles = NNODES / 16;             // 3125
  const int edge_tiles = NEDGES / 16;             // 25000
  node_proj_kernel<<<(node_tiles + 7) / 8, 256, 3 * WBYTES, stream>>>(
      node_feats, WB, bq, bk, bv, Q, K, V);
  edge_kernel<<<(edge_tiles + 7) / 8, 256, 2 * WBYTES, stream>>>(
      edge_feats, src, dst, Q, K, V, WB, be, bev, ln_eg, ln_eb, AGG, Y);
  node_final_kernel<<<(node_tiles + 7) / 8, 256, 1 * WBYTES, stream>>>(
      node_feats, AGG, WB, bout, ln_ng, ln_nb, X);
}
